// FasterRCNN_44890998177909
// MI455X (gfx1250) — matrix operand order repaired and checked by reading, not compiled, not run on hardware
//
#include <hip/hip_runtime.h>

typedef __attribute__((ext_vector_type(8)))  float   v8f;
typedef __attribute__((ext_vector_type(8)))  __bf16  v8bf;
typedef __attribute__((ext_vector_type(16))) __bf16  v16bf;

#define N_PTS    8192
#define F_DIM    128
#define MARGIN   0.2f
#define NEG_FILL 1e30f

// Tile config: block = 256 threads = 8 waves, tile = 64 rows x 128 cols, K-chunk 64
#define BM  64
#define BN  128
#define KB  64
#define LDR 72    // bf16 leading dim, row tile  (144B, 16B-aligned, non-pow2 -> bank spread)
#define LDC 72    // bf16 leading dim, col tile

// ---------------------------------------------------------------- row norms (exact f32)
__global__ __launch_bounds__(256) void norms_kernel(const float* __restrict__ A,
                                                    float* __restrict__ nrm) {
    int wave = (blockIdx.x * blockDim.x + threadIdx.x) >> 5;
    int lane = threadIdx.x & 31;
    if (wave >= N_PTS) return;
    const float4 v = ((const float4*)(A + (size_t)wave * F_DIM))[lane];
    float s = v.x * v.x + v.y * v.y + v.z * v.z + v.w * v.w;
    #pragma unroll
    for (int off = 16; off > 0; off >>= 1) s += __shfl_xor(s, off, 32);
    if (lane == 0) nrm[wave] = s;
}

// ---------------------------------------------------------------- init mining state
__global__ __launch_bounds__(256) void init_kernel(int* __restrict__ posBits,
                                                   int* __restrict__ negBits) {
    int i = blockIdx.x * blockDim.x + threadIdx.x;
    if (i < N_PTS) {
        posBits[i] = 0;                        // distances >= 0; diag always in "same"
        negBits[i] = __float_as_int(NEG_FILL);
    }
}

// ---------------------------------------------------------------- fused GEMM + mining
__global__ __launch_bounds__(256, 2) void tile_kernel(const float* __restrict__ A,
                                                      const float* __restrict__ nrm,
                                                      const int* __restrict__ ids,
                                                      int* __restrict__ posBits,
                                                      int* __restrict__ negBits) {
    // bf16 hi/lo split tiles, row-major [row][k] so fragments are contiguous b128 loads
    __shared__ __bf16 ldsRH[BM * LDR];
    __shared__ __bf16 ldsRL[BM * LDR];
    __shared__ __bf16 ldsCH[BN * LDC];
    __shared__ __bf16 ldsCL[BN * LDC];

    const int tid  = threadIdx.x;
    const int lane = tid & 31;
    const int wave = tid >> 5;
    const int ll   = lane & 15;   // position within half
    const int lh   = lane >> 4;   // which half
    const int wrow = wave & 3;    // 4 row sub-bands of 16
    const int wcol = wave >> 2;   // 2 col groups of 64
    const int i0   = blockIdx.y * BM;
    const int j0   = blockIdx.x * BN;

    v8f acc[4];
    #pragma unroll
    for (int s = 0; s < 4; ++s)
        #pragma unroll
        for (int e = 0; e < 8; ++e) acc[s][e] = 0.0f;

    const int r = wrow * 16 + ll;            // LDS row index for A-fragment

    for (int k0 = 0; k0 < F_DIM; k0 += KB) {
        __syncthreads();
        // ---- stage + split 64xKB row tile (4 float4/thread)
        #pragma unroll
        for (int e = 0; e < 4; ++e) {
            int slot = tid + e * 256;            // 0..1023 float4 slots
            int rr   = slot >> 4;                // row 0..63
            int k4   = slot & 15;                // float4 index along k
            float4 v = *(const float4*)(A + (size_t)(i0 + rr) * F_DIM + k0 + k4 * 4);
            float f[4] = {v.x, v.y, v.z, v.w};
            #pragma unroll
            for (int j = 0; j < 4; ++j) {
                __bf16 h = (__bf16)f[j];
                __bf16 l = (__bf16)(f[j] - (float)h);
                ldsRH[rr * LDR + k4 * 4 + j] = h;
                ldsRL[rr * LDR + k4 * 4 + j] = l;
            }
        }
        // ---- stage + split 128xKB col tile (8 float4/thread)
        #pragma unroll
        for (int e = 0; e < 8; ++e) {
            int slot = tid + e * 256;            // 0..2047 float4 slots
            int cc   = slot >> 4;                // col 0..127
            int k4   = slot & 15;
            float4 v = *(const float4*)(A + (size_t)(j0 + cc) * F_DIM + k0 + k4 * 4);
            float f[4] = {v.x, v.y, v.z, v.w};
            #pragma unroll
            for (int j = 0; j < 4; ++j) {
                __bf16 h = (__bf16)f[j];
                __bf16 l = (__bf16)(f[j] - (float)h);
                ldsCH[cc * LDC + k4 * 4 + j] = h;
                ldsCL[cc * LDC + k4 * 4 + j] = l;
            }
        }
        __syncthreads();

        // ---- K-chunks of 32: 3x v_wmma_f32_16x16x32_bf16 per subtile (hi*hi+hi*lo+lo*hi)
        #pragma unroll
        for (int kc = 0; kc < KB; kc += 32) {
            // A fragment: lanes0-15 K={kc..kc+7, kc+16..kc+23}; lanes16-31 shifted by 8
            const int ra = r * LDR + kc + lh * 8;
            v8bf ah0 = *(const v8bf*)&ldsRH[ra];
            v8bf ah1 = *(const v8bf*)&ldsRH[ra + 16];
            v8bf al0 = *(const v8bf*)&ldsRL[ra];
            v8bf al1 = *(const v8bf*)&ldsRL[ra + 16];
            v16bf aHi, aLo;
            #pragma unroll
            for (int e = 0; e < 8; ++e) {
                aHi[e] = ah0[e]; aHi[8 + e] = ah1[e];
                aLo[e] = al0[e]; aLo[8 + e] = al1[e];
            }
            #pragma unroll
            for (int s = 0; s < 4; ++s) {
                // B fragment: lane = col n, K = {kc+lh*8..+7, kc+16+lh*8..+7} (matches A)
                const int n  = wcol * 64 + s * 16 + ll;
                const int cb = n * LDC + kc + lh * 8;
                v8bf bh0 = *(const v8bf*)&ldsCH[cb];
                v8bf bh1 = *(const v8bf*)&ldsCH[cb + 16];
                v8bf bl0 = *(const v8bf*)&ldsCL[cb];
                v8bf bl1 = *(const v8bf*)&ldsCL[cb + 16];
                v16bf bHi, bLo;
                #pragma unroll
                for (int e = 0; e < 8; ++e) {
                    bHi[e] = bh0[e]; bHi[8 + e] = bh1[e];
                    bLo[e] = bl0[e]; bLo[8 + e] = bl1[e];
                }
                acc[s] = __builtin_amdgcn_wmma_f32_16x16x32_bf16(
                    false, aHi, false, bHi, (short)0, acc[s], false, false);
                acc[s] = __builtin_amdgcn_wmma_f32_16x16x32_bf16(
                    false, aHi, false, bLo, (short)0, acc[s], false, false);
                acc[s] = __builtin_amdgcn_wmma_f32_16x16x32_bf16(
                    false, aLo, false, bHi, (short)0, acc[s], false, false);
            }
        }
    }

    // ---- epilogue: distances + batch-hard mining (fused; D never hits memory)
    const int rowBase = i0 + wrow * 16;
    int   idi[8];
    float nai[8];
    #pragma unroll
    for (int v = 0; v < 8; ++v) {
        int m  = rowBase + v + 8 * lh;           // C layout: VGPR v, half lh -> row v+8*lh
        idi[v] = ids[m];
        nai[v] = nrm[m];
    }
    float posM[8], negM[8];
    #pragma unroll
    for (int v = 0; v < 8; ++v) { posM[v] = 0.0f; negM[v] = NEG_FILL; }

    #pragma unroll
    for (int s = 0; s < 4; ++s) {
        int   n   = j0 + wcol * 64 + s * 16 + ll;
        int   idj = ids[n];
        float nb  = nrm[n];
        #pragma unroll
        for (int v = 0; v < 8; ++v) {
            float d2 = nai[v] + nb - 2.0f * acc[s][v];
            float d  = sqrtf(fmaxf(d2, 0.0f) + 1e-12f);
            bool same = (idi[v] == idj);
            posM[v] = same ? fmaxf(posM[v], d) : posM[v];
            negM[v] = same ? negM[v] : fminf(negM[v], d);
        }
    }

    // per-row reduce across the 16 lanes of each half (halves hold different rows)
    #pragma unroll
    for (int off = 1; off < 16; off <<= 1) {
        #pragma unroll
        for (int v = 0; v < 8; ++v) {
            posM[v] = fmaxf(posM[v], __shfl_xor(posM[v], off, 32));
            negM[v] = fminf(negM[v], __shfl_xor(negM[v], off, 32));
        }
    }
    if (ll == 0) {   // lanes 0 and 16
        #pragma unroll
        for (int v = 0; v < 8; ++v) {
            int m = rowBase + v + 8 * lh;
            atomicMax(&posBits[m], __float_as_int(posM[v]));  // monotone for floats >= 0
            atomicMin(&negBits[m], __float_as_int(negM[v]));
        }
    }
}

// ---------------------------------------------------------------- final deterministic sum
__global__ __launch_bounds__(256) void reduce_kernel(const int* __restrict__ posBits,
                                                     const int* __restrict__ negBits,
                                                     float* __restrict__ out) {
    __shared__ float red[256];
    float s = 0.0f;
    for (int i = threadIdx.x; i < N_PTS; i += 256) {
        float hp = __int_as_float(posBits[i]);
        float hn = __int_as_float(negBits[i]);   // stays 1e30 if no negatives -> loss 0
        s += fmaxf(MARGIN + hp - hn, 0.0f);
    }
    red[threadIdx.x] = s;
    __syncthreads();
    for (int off = 128; off > 0; off >>= 1) {
        if (threadIdx.x < off) red[threadIdx.x] += red[threadIdx.x + off];
        __syncthreads();
    }
    if (threadIdx.x == 0) out[0] = red[0] / (float)N_PTS;
}

// ---------------------------------------------------------------- launch
extern "C" void kernel_launch(void* const* d_in, const int* in_sizes, int n_in,
                              void* d_out, int out_size, void* d_ws, size_t ws_size,
                              hipStream_t stream) {
    const float* A   = (const float*)d_in[0];   // [N, F] f32
    const int*   ids = (const int*)d_in[1];     // [N] i32
    float* out = (float*)d_out;

    // workspace: posBits[N], negBits[N], nrm[N]
    int*   posBits = (int*)d_ws;
    int*   negBits = posBits + N_PTS;
    float* nrm     = (float*)(negBits + N_PTS);

    init_kernel<<<N_PTS / 256, 256, 0, stream>>>(posBits, negBits);
    norms_kernel<<<N_PTS / 8, 256, 0, stream>>>(A, nrm);   // 8 waves/block, 1 row/wave
    dim3 grid(N_PTS / BN, N_PTS / BM);
    tile_kernel<<<grid, 256, 0, stream>>>(A, nrm, ids, posBits, negBits);
    reduce_kernel<<<1, 256, 0, stream>>>(posBits, negBits, out);
}
